// SpatialGRU_42949672961828
// MI455X (gfx1250) — matrix operand order repaired and checked by reading, not compiled, not run on hardware
//
#include <hip/hip_runtime.h>
#include <hip/hip_bf16.h>

#define BATCH 128
#define CCH   32
#define LROWS 80
#define RCOLS 80
#define UH    128
#define DIN   416            // C + 3U
#define QS    440            // padded LDS row stride in bf16 (880 B, 16B-multiple)
#define BU    (BATCH * UH)
#define RBU   (RCOLS * BU)
#define NDIAG (LROWS + RCOLS - 1)
#define SMEM_BYTES (2 * BATCH * QS * 2)   // q + A2, bf16

typedef __attribute__((ext_vector_type(16))) __bf16 v16bf;
typedef __attribute__((ext_vector_type(8)))  float  v8f;
typedef __attribute__((ext_vector_type(4)))  int    i4;

struct Pair32 { i4 lo, hi; };

// ---- A fragment (16x32 bf16, M x K) from an LDS matrix with row stride QS.
// Lane L: row = mt*16 + (L&15); K chunks [off..off+7] and [off+16..off+23], off = 8*(L>=16).
static __device__ inline v16bf ld_frag_a(const __bf16* base, int mt, int k0, int lane) {
    int row = mt * 16 + (lane & 15);
    int off = (lane >> 4) * 8;
    const __bf16* p = base + row * QS + k0 + off;
    Pair32 pr;
    pr.lo = *(const i4*)(p);
    pr.hi = *(const i4*)(p + 16);
    return __builtin_bit_cast(v16bf, pr);
}

// ---- B fragment (32x16 bf16, K x N) from a row-major weight matrix W[n][k], k-stride DIN.
// Lane L: n = n0 + (L&15); K chunks [off..off+7] and [off+16..off+23], off = k0 + 8*(L>=16).
static __device__ inline v16bf ld_frag_b(const __bf16* W, int n0, int k0, int lane) {
    int row = n0 + (lane & 15);
    int kb  = k0 + (lane >> 4) * 8;
    const __bf16* p = W + row * DIN + kb;
    Pair32 pr;
    pr.lo = *(const i4*)(p);
    pr.hi = *(const i4*)(p + 16);
    return __builtin_bit_cast(v16bf, pr);
}

static __device__ inline v8f wmma_bf16(v16bf a, v16bf b, v8f c) {
    return __builtin_amdgcn_wmma_f32_16x16x32_bf16(false, a, false, b, (short)0, c, false, false);
}

// ---------- weight preprocessing: f32 -> bf16, fused layouts ----------
// Wrz (896 x 416): rows 0..383 = Wr, rows 384..895 = Wz.
// WUij (128 x 416): cols 0..383 = WU, cols 384..415 = Wij.
__global__ void prep_weights(const float* __restrict__ Wr, const float* __restrict__ Wz,
                             const float* __restrict__ WU, const float* __restrict__ Wij,
                             __bf16* __restrict__ Wrz, __bf16* __restrict__ WUij) {
    int idx    = blockIdx.x * blockDim.x + threadIdx.x;
    int stride = gridDim.x * blockDim.x;
    const int totalWrz = (384 + 512) * DIN;
    for (int e = idx; e < totalWrz; e += stride) {
        int n = e / DIN, k = e - n * DIN;
        float v = (n < 384) ? Wr[n * DIN + k] : Wz[(n - 384) * DIN + k];
        Wrz[e] = (__bf16)v;
    }
    const int totalWU = UH * DIN;
    for (int e = idx; e < totalWU; e += stride) {
        int n = e / DIN, k = e - n * DIN;
        float v = (k < 384) ? WU[n * 384 + k] : Wij[n * CCH + (k - 384)];
        WUij[e] = (__bf16)v;
    }
}

// ---------- one cell per workgroup; one diagonal per launch ----------
__global__ __launch_bounds__(1024, 1)
void cell_kernel(const float* __restrict__ inputs,
                 const float* __restrict__ br, const float* __restrict__ bz,
                 const float* __restrict__ bij,
                 const __bf16* __restrict__ Wrz, const __bf16* __restrict__ WUij,
                 const float* __restrict__ Hm1,   // h of diagonal d-1 (parity ring)
                 const float* __restrict__ Hm2,   // h of diagonal d-2
                 float* __restrict__ Hcur,        // h of diagonal d (write)
                 int d, int i0) {
    extern __shared__ __align__(16) __bf16 smem[];
    __bf16* q  = smem;                 // 128 x 440 bf16: [h_top | h_left | h_diag | s_ij]
    __bf16* A2 = smem + BATCH * QS;    // 128 x 440 bf16: [r*h_left | r*h_top | r*h_diag | s_ij]

    const int i = i0 + (int)blockIdx.x;
    const int j = d - i;
    const int tid  = threadIdx.x;
    const int wv   = tid >> 5;
    const int lane = tid & 31;
    const int ln15 = lane & 15;
    const int hi16 = lane >> 4;

    // -------- Phase 1: build q (and copy s into A2) --------
    for (int e = tid; e < BATCH * DIN; e += 1024) {
        int row = e / DIN;
        int col = e - row * DIN;
        float v;
        if (col < 128) {                       // h_top = h(i-1, j)
            v = (i > 0) ? Hm1[(size_t)j * BU + row * UH + col] : 0.0f;
        } else if (col < 256) {                // h_left = h(i, j-1)
            v = (j > 0) ? Hm1[(size_t)(j - 1) * BU + row * UH + (col - 128)] : 0.0f;
        } else if (col < 384) {                // h_diag = h(i-1, j-1)
            v = (i > 0 && j > 0) ? Hm2[(size_t)(j - 1) * BU + row * UH + (col - 256)] : 0.0f;
        } else {                               // s_ij[b][c] = inputs[b][c][i][j]
            int c = col - 384;
            v = inputs[(((size_t)row * CCH + c) * LROWS + i) * RCOLS + j];
        }
        __bf16 bv = (__bf16)v;
        q[row * QS + col] = bv;
        if (col >= 384) A2[row * QS + col] = bv;   // s part of A2 has no r gate
    }
    __syncthreads();

    // -------- Phase 2a: r tiles (192 = 8 m-tiles x 24 n-tiles), 6 per wave --------
    // r = sigmoid(q @ Wr.T + br); A2[:, c] = r[:, c] * hcat[:, c], hcat = [h_left|h_top|h_diag]
    #pragma unroll
    for (int jj = 0; jj < 6; ++jj) {
        int t  = wv * 6 + jj;          // 0..191
        int mt = t / 24;
        int nt = t - mt * 24;
        v8f acc = {0.f, 0.f, 0.f, 0.f, 0.f, 0.f, 0.f, 0.f};
        for (int k0 = 0; k0 < DIN; k0 += 32) {
            v16bf a = ld_frag_a(q, mt, k0, lane);
            v16bf b = ld_frag_b(Wrz, nt * 16, k0, lane);
            acc = wmma_bf16(a, b, acc);
        }
        // r col c -> q col: c<128: +128 (h_left), c<256: -128 (h_top), else 0 (h_diag)
        int   delta = (nt < 8) ? 128 : ((nt < 16) ? -128 : 0);
        int   ccol  = nt * 16 + ln15;
        float bias  = br[ccol];
        #pragma unroll
        for (int r8 = 0; r8 < 8; ++r8) {
            int   mrow = mt * 16 + r8 + 8 * hi16;
            float rg   = 1.0f / (1.0f + __expf(-(acc[r8] + bias)));
            float hv   = (float)q[mrow * QS + ccol + delta];
            A2[mrow * QS + ccol] = (__bf16)(rg * hv);
        }
    }

    // -------- Phase 2b: z tiles -- 64 jobs (8 m x 8 u), 2 per wave, 4 groups in-register --------
    float zi_r[2][8], hp_r[2][8];
    #pragma unroll
    for (int jj = 0; jj < 2; ++jj) {
        int zj = wv * 2 + jj;          // 0..63
        int mt = zj >> 3;
        int ut = zj & 7;
        v8f a0 = {0.f,0.f,0.f,0.f,0.f,0.f,0.f,0.f};
        v8f a1 = a0, a2v = a0, a3 = a0;
        for (int k0 = 0; k0 < DIN; k0 += 32) {
            v16bf a = ld_frag_a(q, mt, k0, lane);        // A reused across 4 groups
            a0  = wmma_bf16(a, ld_frag_b(Wrz, 384 + 0 * 128 + ut * 16, k0, lane), a0);
            a1  = wmma_bf16(a, ld_frag_b(Wrz, 384 + 1 * 128 + ut * 16, k0, lane), a1);
            a2v = wmma_bf16(a, ld_frag_b(Wrz, 384 + 2 * 128 + ut * 16, k0, lane), a2v);
            a3  = wmma_bf16(a, ld_frag_b(Wrz, 384 + 3 * 128 + ut * 16, k0, lane), a3);
        }
        int   u  = ut * 16 + ln15;
        float b0 = bz[u], b1 = bz[128 + u], b2 = bz[256 + u], b3 = bz[384 + u];
        #pragma unroll
        for (int r8 = 0; r8 < 8; ++r8) {
            float l0 = a0[r8] + b0, l1 = a1[r8] + b1, l2 = a2v[r8] + b2, l3 = a3[r8] + b3;
            float mx = fmaxf(fmaxf(l0, l1), fmaxf(l2, l3));
            float e0 = __expf(l0 - mx), e1 = __expf(l1 - mx);
            float e2 = __expf(l2 - mx), e3 = __expf(l3 - mx);
            float inv = 1.0f / (e0 + e1 + e2 + e3);
            int   brow = mt * 16 + r8 + 8 * hi16;
            float hl = (j > 0)          ? Hm1[(size_t)(j - 1) * BU + brow * UH + u] : 0.0f;
            float ht = (i > 0)          ? Hm1[(size_t)j       * BU + brow * UH + u] : 0.0f;
            float hd = (i > 0 && j > 0) ? Hm2[(size_t)(j - 1) * BU + brow * UH + u] : 0.0f;
            // group order: zi, zl, zt, zd
            hp_r[jj][r8] = (e1 * hl + e2 * ht + e3 * hd) * inv;
            zi_r[jj][r8] = e0 * inv;
        }
    }

    __syncthreads();   // A2 fully written by all waves before GEMM2

    // -------- Phase 3: GEMM2 (same (mt,ut) tiles as the wave's z jobs) + combine --------
    #pragma unroll
    for (int jj = 0; jj < 2; ++jj) {
        int zj = wv * 2 + jj;
        int mt = zj >> 3;
        int ut = zj & 7;
        v8f acc = {0.f,0.f,0.f,0.f,0.f,0.f,0.f,0.f};
        for (int k0 = 0; k0 < DIN; k0 += 32) {
            v16bf a = ld_frag_a(A2, mt, k0, lane);
            v16bf b = ld_frag_b(WUij, ut * 16, k0, lane);
            acc = wmma_bf16(a, b, acc);
        }
        int   u  = ut * 16 + ln15;
        float bb = bij[u];
        #pragma unroll
        for (int r8 = 0; r8 < 8; ++r8) {
            float hhat = tanhf(acc[r8] + bb);
            float h    = hp_r[jj][r8] + zi_r[jj][r8] * hhat;
            int   brow = mt * 16 + r8 + 8 * hi16;
            Hcur[(size_t)j * BU + brow * UH + u] = h;
        }
    }
}

__global__ void copy_out(const float* __restrict__ src, float* __restrict__ dst) {
    int idx = blockIdx.x * blockDim.x + threadIdx.x;
    if (idx < BU) dst[idx] = src[idx];
}

extern "C" void kernel_launch(void* const* d_in, const int* in_sizes, int n_in,
                              void* d_out, int out_size, void* d_ws, size_t ws_size,
                              hipStream_t stream) {
    (void)in_sizes; (void)n_in; (void)out_size; (void)ws_size;
    const float* inputs = (const float*)d_in[0];
    const float* Wr     = (const float*)d_in[1];
    const float* br     = (const float*)d_in[2];
    const float* Wz     = (const float*)d_in[3];
    const float* bz     = (const float*)d_in[4];
    const float* Wij    = (const float*)d_in[5];
    const float* bij    = (const float*)d_in[6];
    const float* WU     = (const float*)d_in[7];

    char* ws = (char*)d_ws;
    const size_t offWrz  = 0;
    const size_t szWrz   = (size_t)(384 + 512) * DIN * sizeof(__bf16);   // 745,472 B
    const size_t offWUij = offWrz + szWrz;
    const size_t szWUij  = (size_t)UH * DIN * sizeof(__bf16);            // 106,496 B
    const size_t offH    = offWUij + szWUij;                             // 3-parity h ring

    __bf16* Wrz  = (__bf16*)(ws + offWrz);
    __bf16* WUij = (__bf16*)(ws + offWUij);
    float*  Hb   = (float*)(ws + offH);                                  // 3 * R * B * U f32

    prep_weights<<<512, 256, 0, stream>>>(Wr, Wz, WU, Wij, Wrz, WUij);

    for (int d = 0; d < NDIAG; ++d) {
        int i0     = (d > RCOLS - 1) ? d - (RCOLS - 1) : 0;
        int imax   = (d < LROWS - 1) ? d : LROWS - 1;
        int ncells = imax - i0 + 1;
        float*       Hcur = Hb + (size_t)(d % 3) * RBU;
        const float* Hm1  = Hb + (size_t)((d + 2) % 3) * RBU;   // (d-1) mod 3
        const float* Hm2  = Hb + (size_t)((d + 1) % 3) * RBU;   // (d-2) mod 3
        cell_kernel<<<ncells, 1024, SMEM_BYTES, stream>>>(
            inputs, br, bz, bij, Wrz, WUij, Hm1, Hm2, Hcur, d, i0);
    }

    const float* last = Hb + (size_t)((NDIAG - 1) % 3) * RBU + (size_t)(RCOLS - 1) * BU;
    copy_out<<<(BU + 255) / 256, 256, 0, stream>>>(last, (float*)d_out);
}
